// DeepSeekMoE_4956392259707
// MI455X (gfx1250) — matrix operand order repaired and checked by reading, not compiled, not run on hardware
//
#include <hip/hip_runtime.h>
#include <math.h>

// ---------------------------------------------------------------------------
// DeepSeek-style MoE (dense all-expert reference) for MI455X / gfx1250.
// Compute-bound (~522 GFLOP vs ~180MB traffic) -> bf16 WMMA path:
//   kernel 1: convert x + all weights fp32->bf16 (weights pre-TRANSPOSED so
//             every LDS staging step is a contiguous b128 block copy)
//   kernel 2: router (softmax + top-2 + renorm) -> dense expert weights
//   kernel 3: fused 9-expert SwiGLU GEMM chain on v_wmma_f32_16x16x32_bf16
// 512 threads (16 wave32) per WG; staging via global_load_async_to_lds_b128
// (ASYNCcnt) with division-free, statically-assigned copy indices; k-loops
// use bounded unrolling so WMMA fragments stay in registers.
// ---------------------------------------------------------------------------

typedef __bf16 bf16;
typedef bf16  v16bf __attribute__((ext_vector_type(16)));
typedef bf16  v8bf  __attribute__((ext_vector_type(8)));
typedef float v8f   __attribute__((ext_vector_type(8)));
typedef int   v4i   __attribute__((ext_vector_type(4)));

constexpr int    kB  = 4, kS = 8192;
constexpr int    kT  = kB * kS;          // 32768 tokens
constexpr int    kH  = 576;              // hidden
constexpr int    kI  = 512;              // intermediate
constexpr int    kNE = 9;                // 8 routed + shared (index 8, weight 1.0)
constexpr size_t kHI = (size_t)kH * kI;  // 294912 elements per matrix

constexpr int TM = 64;                   // token tile per workgroup
constexpr int KC = 64;                   // I-chunk width
constexpr int NT = 512;                  // threads per workgroup (16 waves)

// LDS strides (bf16 halfs); multiples of 8 so b128 frags stay 16B-aligned
constexpr int XS_STRIDE = kH;            // x tile   [64][576]
constexpr int WT_STRIDE = kH;            // Wg/Wu^T  [64 n][576 k]
constexpr int DT_STRIDE = KC;            // Wd^T     [576 h][64 r]
constexpr int HB_STRIDE = KC;            // h tile   [64][64]

constexpr int XS_BYTES  = TM * XS_STRIDE * 2;            // 73728
constexpr int WT_BYTES  = TM * WT_STRIDE * 2;            // 73728 (dT is 576*64*2 = same)
constexpr int HB_BYTES  = TM * HB_STRIDE * 2;            // 8192
constexpr int SMEM_BYTES = XS_BYTES + WT_BYTES + HB_BYTES + TM * 4;  // 155904 B

// ---------------------------------------------------------------------------
// CDNA5 async global->LDS staging (ASYNCcnt-tracked path).
// Falls back to a plain b128 LDS store if the builtins are absent.
// ---------------------------------------------------------------------------
#if defined(__AMDGCN__) && __has_builtin(__builtin_amdgcn_global_load_async_to_lds_b128)
#define MOE_ASYNC_LDS 1
#else
#define MOE_ASYNC_LDS 0
#endif

#if MOE_ASYNC_LDS
typedef __attribute__((address_space(1))) v4i glob_v4i;   // HIP "__device__" AS
typedef __attribute__((address_space(3))) v4i lds_v4i;    // HIP "__shared__" AS
#endif

__device__ __forceinline__ void stage_b128(void* lds_dst, const void* gsrc) {
#if MOE_ASYNC_LDS
  __builtin_amdgcn_global_load_async_to_lds_b128(
      (glob_v4i*)gsrc, (lds_v4i*)lds_dst, 0, 0);
#else
  *reinterpret_cast<v8bf*>(lds_dst) = *reinterpret_cast<const v8bf*>(gsrc);
#endif
}

__device__ __forceinline__ void stage_fence() {
#if MOE_ASYNC_LDS
#if __has_builtin(__builtin_amdgcn_s_wait_asynccnt)
  __builtin_amdgcn_s_wait_asynccnt(0);
#else
  asm volatile("s_wait_asynccnt 0x0" ::: "memory");
#endif
#endif
}

// ---------------------------------------------------------------------------
// Fragment helpers.  ISA 7.12.2 layouts (wave32):
//   A 16x32 bf16 : lane m=L%16; lo-half lanes hold K 0..7 & 16..23, hi-half 8..15 & 24..31
//   B 32x16 bf16 : lane n=L%16; lo-half lanes hold K 0..7 & 16..23, hi-half 8..15 & 24..31
//   C 16x16 f32  : VGPR r -> M = r (lanes 0-15) / r+8 (lanes 16-31); N = L%16
// LDS rows are bank-swizzled in 8-half blocks:  phys_blk = logical_blk ^ (row&7)
// ---------------------------------------------------------------------------
__device__ __forceinline__ v16bf frag_join(v8bf lo, v8bf up) {
  return __builtin_shufflevector(lo, up, 0,1,2,3,4,5,6,7,8,9,10,11,12,13,14,15);
}

__device__ __forceinline__ v16bf load_frag_a(const bf16* buf, int stride, int row,
                                             int k0, int hi) {
  const bf16* rp = buf + row * stride;
  const int s  = row & 7;
  const int kb = k0 >> 3;
  v8bf lo = *reinterpret_cast<const v8bf*>(rp + ((((kb + hi)     ^ s) << 3)));
  v8bf up = *reinterpret_cast<const v8bf*>(rp + ((((kb + 2 + hi) ^ s) << 3)));
  return frag_join(lo, up);
}

__device__ __forceinline__ v16bf load_frag_b(const bf16* buf, int stride, int row,
                                             int k0, int hi) {
  const bf16* rp = buf + row * stride;
  const int s  = row & 7;
  const int kb = (k0 >> 3) + hi;
  v8bf lo = *reinterpret_cast<const v8bf*>(rp + (((kb     ^ s) << 3)));
  v8bf up = *reinterpret_cast<const v8bf*>(rp + ((((kb+2) ^ s) << 3)));
  return frag_join(lo, up);
}

__device__ __forceinline__ v8f wmma_bf16(v16bf a, v16bf b, v8f c) {
  return __builtin_amdgcn_wmma_f32_16x16x32_bf16(false, a, false, b, (short)0, c,
                                                 false, false);
}

// one 16x16 tile of (64x576 @ 576x64): 18 k-steps.
// Bounded unroll: at most 2 fragment pairs in flight -> no spills.
__device__ __forceinline__ void gemm_tile(const bf16* xs, const bf16* wt,
                                          int arow, int brow, int hi, v8f& c) {
  #pragma clang loop unroll_count(2)
  for (int kk = 0; kk < kH / 32; ++kk) {
    v16bf a = load_frag_a(xs, XS_STRIDE, arow, kk * 32, hi);
    v16bf b = load_frag_b(wt, WT_STRIDE, brow, kk * 32, hi);
    c = wmma_bf16(a, b, c);
  }
}

// ---------------------------------------------------------------------------
// Kernel 1: fp32 -> bf16 conversion.  Weight layouts are transposed so the
// main kernel's LDS tiles are contiguous 16B blocks in global memory:
//   gbf/ubf : [e][i][h]   (gate/up transposed to [I][H])
//   dbf     : [e][h][i]   (down transposed to [H][I])
// Shared expert is packed as e == 8.
// ---------------------------------------------------------------------------
__global__ void convert_kernel(const float* __restrict__ x,
                               const float* __restrict__ sg, const float* __restrict__ su,
                               const float* __restrict__ sd,
                               const float* __restrict__ rg, const float* __restrict__ ru,
                               const float* __restrict__ rd,
                               bf16* __restrict__ xbf, bf16* __restrict__ gbf,
                               bf16* __restrict__ ubf, bf16* __restrict__ dbf) {
  const size_t i0     = (size_t)blockIdx.x * blockDim.x + threadIdx.x;
  const size_t stride = (size_t)gridDim.x * blockDim.x;
  for (size_t k = i0; k < (size_t)kT * kH; k += stride) xbf[k] = (bf16)x[k];
  for (size_t k = i0; k < (size_t)kNE * kHI; k += stride) {
    const size_t e = k / kHI, r = k - e * kHI;
    // gate/up: dst [i][h] <- src [h][i]
    const size_t gi = r / kH, gh = r - gi * kH;
    float g, u;
    if (e < 8) { g = rg[e * kHI + gh * kI + gi]; u = ru[e * kHI + gh * kI + gi]; }
    else       { g = sg[gh * kI + gi];           u = su[gh * kI + gi]; }
    gbf[k] = (bf16)g; ubf[k] = (bf16)u;
    // down: dst [h][i] <- src [i][h]
    const size_t dh = r / kI, di = r - dh * kI;
    const float d = (e < 8) ? rd[e * kHI + di * kH + dh] : sd[di * kH + dh];
    dbf[k] = (bf16)d;
  }
}

// ---------------------------------------------------------------------------
// Kernel 2: router  (logits -> softmax -> top2 -> renormalized dense weights)
// ---------------------------------------------------------------------------
__global__ void router_kernel(const float* __restrict__ x, const float* __restrict__ rw,
                              const float* __restrict__ rb, float* __restrict__ expw) {
  const int t = blockIdx.x * blockDim.x + threadIdx.x;
  if (t >= kT) return;
  float acc[8];
  #pragma unroll
  for (int e = 0; e < 8; ++e) acc[e] = rb[e];
  const float* xr = x + (size_t)t * kH;
  for (int h = 0; h < kH; ++h) {
    const float xv = xr[h];
    #pragma unroll
    for (int e = 0; e < 8; ++e) acc[e] += xv * rw[h * 8 + e];
  }
  float m = acc[0];
  #pragma unroll
  for (int e = 1; e < 8; ++e) m = fmaxf(m, acc[e]);
  float p[8], s = 0.f;
  #pragma unroll
  for (int e = 0; e < 8; ++e) { p[e] = __expf(acc[e] - m); s += p[e]; }
  const float inv = 1.f / s;
  #pragma unroll
  for (int e = 0; e < 8; ++e) p[e] *= inv;
  int i1 = 0;
  #pragma unroll
  for (int e = 1; e < 8; ++e) if (p[e] > p[i1]) i1 = e;
  int i2 = (i1 == 0) ? 1 : 0;
  #pragma unroll
  for (int e = 0; e < 8; ++e) if (e != i1 && p[e] > p[i2]) i2 = e;
  const float sw = p[i1] + p[i2];
  #pragma unroll
  for (int e = 0; e < 8; ++e)
    expw[(size_t)t * 8 + e] = (e == i1 || e == i2) ? p[e] / sw : 0.f;
}

// ---------------------------------------------------------------------------
// Kernel 3: fused 9-expert SwiGLU MoE.  One WG = 64 tokens, 16 wave32.
// Persistent C: 64x576 f32 spread as 9 tiles/wave. Per (expert, I-chunk):
//   gate/up 64x64 WMMA -> h = w_e*silu(g)*u (bf16 LDS) -> C += h @ Wd_chunk.
// ---------------------------------------------------------------------------
__global__ __launch_bounds__(NT, 1)
void moe_fused_kernel(const bf16* __restrict__ xbf, const bf16* __restrict__ gbf,
                      const bf16* __restrict__ ubf, const bf16* __restrict__ dbf,
                      const float* __restrict__ expw, float* __restrict__ out) {
  extern __shared__ char smem[];
  bf16*  xs  = reinterpret_cast<bf16*>(smem);
  bf16*  wt  = reinterpret_cast<bf16*>(smem + XS_BYTES);   // Wg/Wu^T, reused as Wd^T
  bf16*  hb  = reinterpret_cast<bf16*>(smem + XS_BYTES + WT_BYTES);
  float* wsm = reinterpret_cast<float*>(smem + XS_BYTES + WT_BYTES + HB_BYTES);

  const int tid    = threadIdx.x;
  const int lane   = tid & 31;
  const int wave   = tid >> 5;            // 0..15
  const int hi     = (lane >> 4) & 1;
  const int ln     = lane & 15;
  const int tb     = blockIdx.x * TM;
  const int rowblk = wave >> 2;           // 0..3  (16-token row block)
  const int colA   = wave & 3;            // phase-A col block (of 4)
  const int cbaseB = (wave & 3) * 9;      // phase-B col blocks cbaseB..cbaseB+8 of 36
  const int arow   = rowblk * 16 + ln;    // A-frag row for this wave

  // Division-free staging assignment: 64 rows x 72 blocks = 512 threads x 9.
  const int srow = tid >> 3;              // 0..63  (row this thread copies)
  const int sc0  = tid & 7;               // 0..7   (first block column)
  const int ssw  = srow & 7;              // swizzle for this row
  bf16* const xs_row = xs + srow * XS_STRIDE;
  bf16* const wt_row = wt + srow * WT_STRIDE;

  // ---- stage x tile (bf16, swizzled blocks, contiguous b128 in global) ---
  {
    const bf16* gx = xbf + (size_t)(tb + srow) * kH;
    #pragma unroll
    for (int j = 0; j < 9; ++j) {
      const int c = sc0 + 8 * j;
      stage_b128(xs_row + ((c ^ ssw) << 3), gx + c * 8);
    }
  }

  v8f acc[9];
  const v8f vzero = {0.f, 0.f, 0.f, 0.f, 0.f, 0.f, 0.f, 0.f};
  #pragma unroll
  for (int j = 0; j < 9; ++j) acc[j] = vzero;

  stage_fence();
  for (int e = 0; e < kNE; ++e) {
    if (tid < TM) wsm[tid] = (e < 8) ? expw[(size_t)(tb + tid) * 8 + e] : 1.0f;
    __syncthreads();  // also fences initial xs staging

    const bf16* wg = gbf + (size_t)e * kHI;   // [I][H]
    const bf16* wu = ubf + (size_t)e * kHI;   // [I][H]
    const bf16* wd = dbf + (size_t)e * kHI;   // [H][I]

    for (int ic = 0; ic < kI; ic += KC) {
      // -- stage gate chunk: wt[n][k] = wg[ic+n][k]  (contiguous rows) -----
      {
        const bf16* gsrc = wg + (size_t)(ic + srow) * kH;
        #pragma unroll
        for (int j = 0; j < 9; ++j) {
          const int c = sc0 + 8 * j;
          stage_b128(wt_row + ((c ^ ssw) << 3), gsrc + c * 8);
        }
      }
      // nudge the up-chunk toward L2 while the gate GEMM runs
      if (tid < TM) __builtin_prefetch(&wu[(size_t)(ic + tid) * kH], 0, 1);
      stage_fence();
      __syncthreads();

      v8f g0 = vzero;
      gemm_tile(xs, wt, arow, colA * 16 + ln, hi, g0);
      __syncthreads();   // everyone done reading wt

      // -- stage up chunk --------------------------------------------------
      {
        const bf16* gsrc = wu + (size_t)(ic + srow) * kH;
        #pragma unroll
        for (int j = 0; j < 9; ++j) {
          const int c = sc0 + 8 * j;
          stage_b128(wt_row + ((c ^ ssw) << 3), gsrc + c * 8);
        }
      }
      stage_fence();
      __syncthreads();

      v8f u0 = vzero;
      gemm_tile(xs, wt, arow, colA * 16 + ln, hi, u0);

      // -- h = w_e * silu(g) * u  -> bf16 LDS ------------------------------
      #pragma unroll
      for (int r = 0; r < 8; ++r) {
        const int rt = rowblk * 16 + r + hi * 8;   // token row in tile
        const float w = wsm[rt];
        float s0 = g0[r];  s0 = s0 / (1.0f + __expf(-s0));
        const float h0 = w * s0 * u0[r];
        const int c0 = colA * 16 + ln;
        hb[rt * HB_STRIDE + ((((c0 >> 3) ^ (rt & 7)) << 3) | (c0 & 7))] = (bf16)h0;
      }
      __syncthreads();   // hb complete, wt free

      // -- stage down chunk: dT[h][r] = wd[h][ic+r]  (576 rows x 8 blocks) -
      bf16* dT = wt;
      {
        #pragma unroll
        for (int j = 0; j < 9; ++j) {
          const int idx = tid + NT * j;     // 0..4607
          const int h = idx >> 3, c = idx & 7;
          stage_b128(dT + h * DT_STRIDE + ((c ^ (h & 7)) << 3),
                     wd + (size_t)h * kI + ic + c * 8);
        }
      }
      stage_fence();
      __syncthreads();

      // -- phase B: acc += h(64x64) @ Wd_chunk(64x576) ---------------------
      // (full unroll required: acc[j] must stay register-indexed)
      #pragma unroll
      for (int kk = 0; kk < 2; ++kk) {
        v16bf a = load_frag_a(hb, HB_STRIDE, arow, kk * 32, hi);
        #pragma unroll
        for (int j = 0; j < 9; ++j) {
          v16bf b = load_frag_b(dT, DT_STRIDE, (cbaseB + j) * 16 + ln, kk * 32, hi);
          acc[j] = wmma_bf16(a, b, acc[j]);
        }
      }
      __syncthreads();   // before next chunk restages wt/hb
    }
  }

  // ---- epilogue: each (row, col) written exactly once -------------------
  #pragma unroll
  for (int j = 0; j < 9; ++j) {
    const int col = (cbaseB + j) * 16 + ln;
    #pragma unroll
    for (int r = 0; r < 8; ++r) {
      const int row = tb + rowblk * 16 + r + hi * 8;
      out[(size_t)row * kH + col] = acc[j][r];
    }
  }
}

// ---------------------------------------------------------------------------
// Host launcher.  Workspace layout (needs ~54.7 MB):
//   xbf [T*H] bf16 | gbf [9*I*H] bf16 | ubf [9*I*H] bf16 | dbf [9*H*I] bf16
//   expw [T*8] f32
// ---------------------------------------------------------------------------
extern "C" void kernel_launch(void* const* d_in, const int* in_sizes, int n_in,
                              void* d_out, int out_size, void* d_ws, size_t ws_size,
                              hipStream_t stream) {
  (void)in_sizes; (void)n_in; (void)out_size; (void)ws_size;
  const float* x  = (const float*)d_in[0];
  const float* sg = (const float*)d_in[1];
  const float* su = (const float*)d_in[2];
  const float* sd = (const float*)d_in[3];
  const float* rg = (const float*)d_in[4];
  const float* ru = (const float*)d_in[5];
  const float* rd = (const float*)d_in[6];
  const float* rw = (const float*)d_in[7];
  const float* rb = (const float*)d_in[8];
  float* out = (float*)d_out;

  bf16*  xbf  = (bf16*)d_ws;
  bf16*  gbf  = xbf + (size_t)kT * kH;
  bf16*  ubf  = gbf + (size_t)kNE * kHI;
  bf16*  dbf  = ubf + (size_t)kNE * kHI;
  float* expw = (float*)(dbf + (size_t)kNE * kHI);

  convert_kernel<<<2048, 256, 0, stream>>>(x, sg, su, sd, rg, ru, rd,
                                           xbf, gbf, ubf, dbf);
  router_kernel<<<kT / 256, 256, 0, stream>>>(x, rw, rb, expw);

  (void)hipFuncSetAttribute((const void*)moe_fused_kernel,
                            hipFuncAttributeMaxDynamicSharedMemorySize, SMEM_BYTES);
  moe_fused_kernel<<<kT / TM, NT, SMEM_BYTES, stream>>>(xbf, gbf, ubf, dbf, expw, out);
}
